// STGCN3DGEPModel_66597762892191
// MI455X (gfx1250) — matrix operand order repaired and checked by reading, not compiled, not run on hardware
//
#include <hip/hip_runtime.h>
#include <hip/hip_bf16.h>

typedef __attribute__((ext_vector_type(16))) __bf16 v16bf;
typedef __attribute__((ext_vector_type(8)))  __bf16 v8bf;
typedef __attribute__((ext_vector_type(8)))  float  v8f;

#define NB   128        // batch N
#define VB   128        // V
#define CC   256        // input channels
#define HH   512        // hidden
#define TT   12         // time steps
#define NCLS 5
#define OD   5
#define EHD  256
#define ECD  256
#define MROW (NB*VB)    // 16384 rows
#define G4   (4*HH)     // 2048 gate cols

// ---------- device helpers ----------
__device__ __forceinline__ float sigmoid_f(float x) {
  return 1.0f / (1.0f + __expf(-x));
}
__device__ __forceinline__ float tanh_f(float x) {
  float ax = fabsf(x);
  float e  = __expf(2.0f * ax);
  float t  = 1.0f - 2.0f / (e + 1.0f);
  return copysignf(t, x);
}
__device__ __forceinline__ v8f wmma_bf16(v16bf a, v16bf b, v8f c) {
  // D = A(16x32) * B(32x16) + C, f32 accumulate
  return __builtin_amdgcn_wmma_f32_16x16x32_bf16(false, a, false, b, (short)0, c,
                                                 false, false);
}

// Load A fragment (16x32 bf16, contiguous-k order, same as B) from row-major A[row][K]:
// lane<16: row=rbase+lane, K = kk+0..15; lane>=16: row=rbase+lane-16, K = kk+16..31
struct AFrag { union { v16bf v; v8bf h[2]; } u; };
__device__ __forceinline__ v16bf load_afrag(const __bf16* __restrict__ abase, int kk) {
  AFrag a;
  a.u.h[0] = *(const v8bf*)(abase + kk);
  a.u.h[1] = *(const v8bf*)(abase + kk + 8);
  return a.u.v;
}
// Load B fragment (32x16 bf16 = Wᵀ tile) from row-major W[n][K]:
// lane column n = nbase+(lane&15); lane<16 holds K=kk+0..15, lane>=16 holds K=kk+16..31
__device__ __forceinline__ v16bf load_bfrag(const __bf16* __restrict__ bbase, int kk) {
  return *(const v16bf*)(bbase + kk);
}

// ---------- conversion / init ----------
__global__ void cvt_bf16(const float* __restrict__ in, __bf16* __restrict__ out, int n) {
  int i = blockIdx.x * blockDim.x + threadIdx.x;
  if (i < n) out[i] = (__bf16)in[i];
}
__global__ void init_state(const float* __restrict__ hid, const float* __restrict__ cell,
                           float* __restrict__ h32, float* __restrict__ c32,
                           __bf16* __restrict__ hbf, int n) {
  int i = blockIdx.x * blockDim.x + threadIdx.x;
  if (i < n) {
    float h = hid[i];
    h32[i] = h; c32[i] = cell[i]; hbf[i] = (__bf16)h;
  }
}

// ---------- generic WMMA GEMM: C[M,ldc] = A[M,K](bf16) * B[ldc,K]^T(bf16) + bias ----------
// grid: (M/64, ldc/256), block 256 (8 waves).
// Wave tile: 32 rows x 64 cols  (2 row-frags x 4 col-frags, 8 accumulators).
// Per K-block: 2 A-frags + 4 B-frags feed 8 WMMAs (A reused 4x, B reused 2x).
__global__ __launch_bounds__(256)
void gemm_bias(const __bf16* __restrict__ A, const __bf16* __restrict__ B,
               const float* __restrict__ bias, float* __restrict__ C,
               int K, int ldc) {
  const int lane = threadIdx.x & 31;
  const int wave = threadIdx.x >> 5;
  const int lh = lane & 15, sel = lane >> 4;
  const int rbase = blockIdx.x * 64 + (wave & 1) * 32;
  const int nbase = blockIdx.y * 256 + (wave >> 1) * 64;
  const __bf16* a0 = A + (size_t)(rbase + lh) * K + sel * 16;
  const __bf16* a1 = a0 + (size_t)16 * K;
  const __bf16* bb = B + (size_t)(nbase + lh) * K + sel * 16;
  const size_t bstep = (size_t)16 * K;       // next 16-col frag
  v8f acc[2][4] = {};
#pragma unroll 2
  for (int kk = 0; kk < K; kk += 32) {
    v16bf A0 = load_afrag(a0, kk);
    v16bf A1 = load_afrag(a1, kk);
    v16bf B0 = load_bfrag(bb,             kk);
    v16bf B1 = load_bfrag(bb + bstep,     kk);
    v16bf B2 = load_bfrag(bb + 2 * bstep, kk);
    v16bf B3 = load_bfrag(bb + 3 * bstep, kk);
    acc[0][0] = wmma_bf16(A0, B0, acc[0][0]);
    acc[1][0] = wmma_bf16(A1, B0, acc[1][0]);
    acc[0][1] = wmma_bf16(A0, B1, acc[0][1]);
    acc[1][1] = wmma_bf16(A1, B1, acc[1][1]);
    acc[0][2] = wmma_bf16(A0, B2, acc[0][2]);
    acc[1][2] = wmma_bf16(A1, B2, acc[1][2]);
    acc[0][3] = wmma_bf16(A0, B3, acc[0][3]);
    acc[1][3] = wmma_bf16(A1, B3, acc[1][3]);
  }
#pragma unroll
  for (int cf = 0; cf < 4; ++cf) {
    const int col = nbase + cf * 16 + lh;
    const float bv = bias[col];
#pragma unroll
    for (int rf = 0; rf < 2; ++rf) {
#pragma unroll
      for (int v = 0; v < 8; ++v) {
        int m = rbase + rf * 16 + v + sel * 8;
        C[(size_t)m * ldc + col] = acc[rf][cf][v] + bv;
      }
    }
  }
}

// ---------- fused LSTM gates + pointwise ----------
// gates = gx + h@W_hh^T + b_hh; all four gate quadrants live in registers, then
// c' = sig(f)*c + sig(i)*tanh(g); h' = sig(o)*tanh(c'); writes h32/c32/hbf_out.
// grid: (MROW/64, HH/64), block 256.
// Wave tile: 32 rows x 16 j x 4 quadrants (8 accumulators); A reused 4x, B reused 2x.
__global__ __launch_bounds__(256)
void lstm_gates(const __bf16* __restrict__ hbf_in, const __bf16* __restrict__ Whh,
                const float* __restrict__ gx, const float* __restrict__ bhh,
                float* __restrict__ c32, float* __restrict__ h32,
                __bf16* __restrict__ hbf_out) {
  const int lane = threadIdx.x & 31;
  const int wave = threadIdx.x >> 5;
  const int lh = lane & 15, sel = lane >> 4;
  const int rbase = blockIdx.x * 64 + (wave & 1) * 32;   // 32 rows per wave
  const int jbase = blockIdx.y * 64 + (wave >> 1) * 16;  // 16 hidden cols per wave
  const __bf16* a0 = hbf_in + (size_t)(rbase + lh) * HH + sel * 16;
  const __bf16* a1 = a0 + (size_t)16 * HH;
  const __bf16* b0 = Whh + (size_t)(jbase + lh) * HH + sel * 16;
  v8f acc[2][4] = {};
#pragma unroll 2
  for (int kk = 0; kk < HH; kk += 32) {
    v16bf A0 = load_afrag(a0, kk);
    v16bf A1 = load_afrag(a1, kk);
    v16bf Bi = load_bfrag(b0,                       kk);
    v16bf Bf = load_bfrag(b0 + (size_t) 512 * HH,   kk);
    v16bf Bg = load_bfrag(b0 + (size_t)1024 * HH,   kk);
    v16bf Bo = load_bfrag(b0 + (size_t)1536 * HH,   kk);
    acc[0][0] = wmma_bf16(A0, Bi, acc[0][0]);
    acc[1][0] = wmma_bf16(A1, Bi, acc[1][0]);
    acc[0][1] = wmma_bf16(A0, Bf, acc[0][1]);
    acc[1][1] = wmma_bf16(A1, Bf, acc[1][1]);
    acc[0][2] = wmma_bf16(A0, Bg, acc[0][2]);
    acc[1][2] = wmma_bf16(A1, Bg, acc[1][2]);
    acc[0][3] = wmma_bf16(A0, Bo, acc[0][3]);
    acc[1][3] = wmma_bf16(A1, Bo, acc[1][3]);
  }
  const int col = jbase + lh;
  const float bi_ = bhh[col];
  const float bf_ = bhh[col + 512];
  const float bg_ = bhh[col + 1024];
  const float bo_ = bhh[col + 1536];
#pragma unroll
  for (int rf = 0; rf < 2; ++rf) {
#pragma unroll
    for (int v = 0; v < 8; ++v) {
      int m = rbase + rf * 16 + v + sel * 8;
      const float* gxr = gx + (size_t)m * G4 + col;
      float gi = acc[rf][0][v] + gxr[0]    + bi_;
      float gf = acc[rf][1][v] + gxr[512]  + bf_;
      float gg = acc[rf][2][v] + gxr[1024] + bg_;
      float go = acc[rf][3][v] + gxr[1536] + bo_;
      size_t idx = (size_t)m * HH + col;
      float cold = c32[idx];
      float cn = sigmoid_f(gf) * cold + sigmoid_f(gi) * tanh_f(gg);
      float hn = sigmoid_f(go) * tanh_f(cn);
      c32[idx] = cn;
      h32[idx] = hn;
      hbf_out[idx] = (__bf16)hn;
    }
  }
}

// ---------- classifier: oc[n] = mean_V(h) @ W_cls^T + b_cls ----------
__global__ __launch_bounds__(256)
void classifier(const float* __restrict__ h32, const float* __restrict__ W_cls,
                const float* __restrict__ b_cls, float* __restrict__ outC, int t) {
  __shared__ float sh[HH];
  const int n = blockIdx.x;
  for (int j = threadIdx.x; j < HH; j += 256) {
    float s = 0.f;
    const float* hp = h32 + (size_t)n * VB * HH + j;
    for (int v = 0; v < VB; ++v) s += hp[(size_t)v * HH];
    sh[j] = s;
  }
  __syncthreads();
  const int wave = threadIdx.x >> 5, lane = threadIdx.x & 31;
  if (wave < NCLS) {
    float s = 0.f;
    for (int j = lane; j < HH; j += 32) s += sh[j] * W_cls[wave * HH + j];
    for (int off = 16; off; off >>= 1) s += __shfl_xor(s, off, 32);
    if (lane == 0)
      outC[((size_t)t * NB + n) * NCLS + wave] = s * (1.0f / VB) + b_cls[wave];
  }
}

// ---------- BN stats for eh (deterministic two-stage) ----------
__global__ __launch_bounds__(256)
void eh_partial(const float* __restrict__ Z, float* __restrict__ part) {
  const int ch = threadIdx.x;          // 256 channels
  const int r0 = blockIdx.x * 128;     // 128 blocks x 128 rows
  float s1 = 0.f, s2 = 0.f;
  for (int r = 0; r < 128; ++r) {
    float z = Z[(size_t)(r0 + r) * EHD + ch];
    s1 += z; s2 += z * z;
  }
  part[blockIdx.x * 512 + ch] = s1;
  part[blockIdx.x * 512 + 256 + ch] = s2;
}
__global__ void eh_finalize(const float* __restrict__ part, float* __restrict__ mean,
                            float* __restrict__ rstd) {
  const int ch = threadIdx.x;
  float s1 = 0.f, s2 = 0.f;
  for (int b = 0; b < 128; ++b) { s1 += part[b * 512 + ch]; s2 += part[b * 512 + 256 + ch]; }
  float m = s1 * (1.0f / MROW);
  float v = s2 * (1.0f / MROW) - m * m;
  mean[ch] = m; rstd[ch] = rsqrtf(v + 1e-5f);
}

// ---------- ec path: one-hot projection + stats ----------
__global__ void ec_compute(const float* __restrict__ oh_t, const float* __restrict__ W_ec,
                           const float* __restrict__ b_ec, float* __restrict__ ecPre) {
  const int n = blockIdx.x, ch = threadIdx.x;
  const float* oh = oh_t + n * NCLS;
  float s = b_ec[ch];
#pragma unroll
  for (int k = 0; k < NCLS; ++k) s += oh[k] * W_ec[ch * NCLS + k];
  ecPre[n * ECD + ch] = s;
}
__global__ void ec_stats(const float* __restrict__ ecPre, float* __restrict__ mean,
                         float* __restrict__ rstd) {
  const int ch = threadIdx.x;
  float s1 = 0.f, s2 = 0.f;
  for (int n = 0; n < NB; ++n) { float z = ecPre[n * ECD + ch]; s1 += z; s2 += z * z; }
  float m = s1 * (1.0f / NB);
  float v = s2 * (1.0f / NB) - m * m;
  mean[ch] = m; rstd[ch] = rsqrtf(v + 1e-5f);
}

// ---------- final: BN+relu, concat, 5-wide GEMM, gaussian activation ----------
__global__ __launch_bounds__(256)
void op_kernel(const float* __restrict__ Z, const float* __restrict__ ecPre,
               const float* __restrict__ ehMean, const float* __restrict__ ehRstd,
               const float* __restrict__ g_eh, const float* __restrict__ be_eh,
               const float* __restrict__ ecMean, const float* __restrict__ ecRstd,
               const float* __restrict__ g_ec, const float* __restrict__ be_ec,
               const float* __restrict__ W_out, const float* __restrict__ b_out,
               float* __restrict__ outPred, int t) {
  const int lane = threadIdx.x & 31, wave = threadIdx.x >> 5;
  const int wg = blockIdx.x * 8 + wave;   // 4096 waves, 4 rows each
  for (int rr = 0; rr < 4; ++rr) {
    int row = wg * 4 + rr;
    int n = row >> 7, vv = row & 127;
    float p0 = 0.f, p1 = 0.f, p2 = 0.f, p3 = 0.f, p4 = 0.f;
#pragma unroll
    for (int c8 = 0; c8 < 8; ++c8) {
      int ch = lane + c8 * 32;
      float z  = Z[(size_t)row * EHD + ch];
      float eh = fmaxf(0.f, g_eh[ch] * (z - ehMean[ch]) * ehRstd[ch] + be_eh[ch]);
      float zc = ecPre[n * ECD + ch];
      float ec = fmaxf(0.f, g_ec[ch] * (zc - ecMean[ch]) * ecRstd[ch] + be_ec[ch]);
      p0 += eh * W_out[0 * 512 + ch] + ec * W_out[0 * 512 + 256 + ch];
      p1 += eh * W_out[1 * 512 + ch] + ec * W_out[1 * 512 + 256 + ch];
      p2 += eh * W_out[2 * 512 + ch] + ec * W_out[2 * 512 + 256 + ch];
      p3 += eh * W_out[3 * 512 + ch] + ec * W_out[3 * 512 + 256 + ch];
      p4 += eh * W_out[4 * 512 + ch] + ec * W_out[4 * 512 + 256 + ch];
    }
    for (int off = 16; off; off >>= 1) {
      p0 += __shfl_xor(p0, off, 32); p1 += __shfl_xor(p1, off, 32);
      p2 += __shfl_xor(p2, off, 32); p3 += __shfl_xor(p3, off, 32);
      p4 += __shfl_xor(p4, off, 32);
    }
    if (lane == 0) {
      size_t base = (((size_t)n * TT + t) * VB + vv) * OD;
      outPred[base + 0] = p0 + b_out[0];
      outPred[base + 1] = p1 + b_out[1];
      outPred[base + 2] = __expf(p2 + b_out[2]);
      outPred[base + 3] = __expf(p3 + b_out[3]);
      outPred[base + 4] = tanh_f(p4 + b_out[4]);
    }
  }
}

// ---------- host orchestration ----------
extern "C" void kernel_launch(void* const* d_in, const int* in_sizes, int n_in,
                              void* d_out, int out_size, void* d_ws, size_t ws_size,
                              hipStream_t stream) {
  const float* x     = (const float*)d_in[0];
  const float* hid0  = (const float*)d_in[1];
  const float* cell0 = (const float*)d_in[2];
  const float* oneh  = (const float*)d_in[3];
  const float* W_ih  = (const float*)d_in[4];
  const float* W_hh  = (const float*)d_in[5];
  const float* b_ih  = (const float*)d_in[6];
  const float* b_hh  = (const float*)d_in[7];
  const float* W_cls = (const float*)d_in[8];
  const float* b_cls = (const float*)d_in[9];
  const float* W_eh  = (const float*)d_in[10];
  const float* b_eh  = (const float*)d_in[11];
  const float* g_eh  = (const float*)d_in[12];
  const float* be_eh = (const float*)d_in[13];
  const float* W_ec  = (const float*)d_in[14];
  const float* b_ec  = (const float*)d_in[15];
  const float* g_ec  = (const float*)d_in[16];
  const float* be_ec = (const float*)d_in[17];
  const float* W_out = (const float*)d_in[18];
  const float* b_out = (const float*)d_in[19];

  char* ws = (char*)d_ws;
  size_t off = 0;
  auto alloc = [&](size_t bytes) -> char* {
    char* p = ws + off;
    off += (bytes + 255) & ~(size_t)255;
    return p;
  };
  float*  gx     = (float*) alloc((size_t)MROW * G4 * 4);   // 134 MB
  float*  h32    = (float*) alloc((size_t)MROW * HH * 4);
  float*  c32    = (float*) alloc((size_t)MROW * HH * 4);
  __bf16* hbf0   = (__bf16*)alloc((size_t)MROW * HH * 2);
  __bf16* hbf1   = (__bf16*)alloc((size_t)MROW * HH * 2);
  __bf16* xbf    = (__bf16*)alloc((size_t)MROW * CC * 2);
  __bf16* Wihb   = (__bf16*)alloc((size_t)G4 * CC * 2);
  __bf16* Whhb   = (__bf16*)alloc((size_t)G4 * HH * 2);
  __bf16* Wehb   = (__bf16*)alloc((size_t)EHD * HH * 2);
  float*  Z      = (float*) alloc((size_t)MROW * EHD * 4);
  float*  ehPart = (float*) alloc((size_t)128 * 512 * 4);
  float*  ehMean = (float*) alloc(256 * 4);
  float*  ehRstd = (float*) alloc(256 * 4);
  float*  ecPre  = (float*) alloc((size_t)NB * ECD * 4);
  float*  ecMean = (float*) alloc(256 * 4);
  float*  ecRstd = (float*) alloc(256 * 4);

  float* outPred = (float*)d_out;
  float* outC    = outPred + (size_t)NB * TT * VB * OD;

  // one-time conversions + state init
  cvt_bf16<<<(MROW * CC + 255) / 256, 256, 0, stream>>>(x, xbf, MROW * CC);
  cvt_bf16<<<(G4 * CC + 255) / 256, 256, 0, stream>>>(W_ih, Wihb, G4 * CC);
  cvt_bf16<<<(G4 * HH + 255) / 256, 256, 0, stream>>>(W_hh, Whhb, G4 * HH);
  cvt_bf16<<<(EHD * HH + 255) / 256, 256, 0, stream>>>(W_eh, Wehb, EHD * HH);
  init_state<<<(MROW * HH + 255) / 256, 256, 0, stream>>>(hid0, cell0, h32, c32, hbf0, MROW * HH);

  // gx = x @ W_ih^T + b_ih  (b_hh is added inside the gates kernel)
  gemm_bias<<<dim3(MROW / 64, G4 / 256), 256, 0, stream>>>(xbf, Wihb, b_ih, gx, CC, G4);

  __bf16* hin = hbf0; __bf16* hout = hbf1;
  for (int t = 0; t < TT; ++t) {
    lstm_gates<<<dim3(MROW / 64, HH / 64), 256, 0, stream>>>(hin, Whhb, gx, b_hh, c32, h32, hout);
    classifier<<<NB, 256, 0, stream>>>(h32, W_cls, b_cls, outC, t);
    gemm_bias<<<dim3(MROW / 64, EHD / 256), 256, 0, stream>>>(hout, Wehb, b_eh, Z, HH, EHD);
    eh_partial<<<128, 256, 0, stream>>>(Z, ehPart);
    eh_finalize<<<1, 256, 0, stream>>>(ehPart, ehMean, ehRstd);
    ec_compute<<<NB, 256, 0, stream>>>(oneh + (size_t)t * NB * NCLS, W_ec, b_ec, ecPre);
    ec_stats<<<1, 256, 0, stream>>>(ecPre, ecMean, ecRstd);
    op_kernel<<<512, 256, 0, stream>>>(Z, ecPre, ehMean, ehRstd, g_eh, be_eh,
                                       ecMean, ecRstd, g_ec, be_ec, W_out, b_out, outPred, t);
    __bf16* tmp = hin; hin = hout; hout = tmp;
  }
}
